// HierarchicalHeteroGraphSage_7301444403985
// MI455X (gfx1250) — matrix operand order repaired and checked by reading, not compiled, not run on hardware
//
#include <hip/hip_runtime.h>

// ---- problem constants (match reference) ----
#define F_IN    128
#define HID     256
#define OUT_F   64
#define NEDGE   500000
#define NP_ALLC 304096
#define NP_H01C 104096
#define NA_ALLC 154096
#define NA_H01C 54096

typedef __attribute__((ext_vector_type(16))) _Float16 v16h;
typedef __attribute__((ext_vector_type(8)))  _Float16 v8h;
typedef __attribute__((ext_vector_type(8)))  float    v8f;

// -------------------- utility kernels --------------------

__global__ void fill0_kernel(float* __restrict__ p, long long n) {
  long long i = (long long)blockIdx.x * blockDim.x + threadIdx.x;
  long long stride = (long long)gridDim.x * blockDim.x;
  for (; i < n; i += stride) p[i] = 0.0f;
}

// W: K x N (f32, row-major)  ->  Wt: N x K (f16, row-major)
__global__ void convert_weight_kernel(const float* __restrict__ W,
                                      _Float16* __restrict__ Wt, int K, int N) {
  int i = blockIdx.x * blockDim.x + threadIdx.x;
  if (i < K * N) {
    int k = i / N, n = i - k * N;
    Wt[(size_t)n * K + k] = (_Float16)W[i];
  }
}

// wave-per-edge segment-sum with destination slicing (dst >= dstLimit is dead work)
__global__ __launch_bounds__(256) void scatter_sum_kernel(
    const float* __restrict__ X, const int* __restrict__ esrc,
    const int* __restrict__ edst, int E, int F, int dstLimit,
    float* __restrict__ sum, float* __restrict__ cnt) {
  int wid  = (int)(((long long)blockIdx.x * blockDim.x + threadIdx.x) >> 5);
  int lane = threadIdx.x & 31;
  int nw   = (int)(((long long)gridDim.x * blockDim.x) >> 5);
  for (int e = wid; e < E; e += nw) {
    int d = edst[e];
    if (d >= dstLimit) continue;
    int s = esrc[e];
    const float* xs = X + (size_t)s * F;
    float* sd = sum + (size_t)d * F;
    for (int f = lane; f < F; f += 32) atomicAdd(&sd[f], xs[f]);
    if (lane == 0) atomicAdd(&cnt[d], 1.0f);
  }
}

__global__ void mean_div_kernel(float* __restrict__ sum, const float* __restrict__ cnt,
                                long long M, int F) {
  long long i = (long long)blockIdx.x * blockDim.x + threadIdx.x;
  if (i < M * (long long)F) {
    long long r = i / F;
    sum[i] = sum[i] / fmaxf(cnt[r], 1.0f);
  }
}

// -------------------- fused multi-pair WMMA GEMM --------------------
// C[M,N] = relu?( sum_p A_p[M,K] @ Bt_p[N,K]^T + bias0 + bias1 )
// All partial products accumulate in WMMA registers; C is written exactly once.
// Block tile: 128(M) x 64(N), 8 waves; each wave: one 16-row subtile, 4 16x16 accs.

struct GemmArgs {
  const float*    A[4];
  const _Float16* Bt[4];
};

__device__ inline v16h load_a_frag(const _Float16* rowp, int khalf) {
  // A-fragment (16-bit A 16x32): lane half h holds K {16h..16h+15}
  v8h lo = *(const v8h*)(rowp + khalf);
  v8h hi = *(const v8h*)(rowp + khalf + 8);
  v16h f;
#pragma unroll
  for (int i = 0; i < 8; ++i) { f[i] = lo[i]; f[i + 8] = hi[i]; }
  return f;
}

__global__ __launch_bounds__(256) void gemm_wmma_fused_kernel(
    GemmArgs args, int nPairs, float* __restrict__ C,
    const float* __restrict__ bias0, const float* __restrict__ bias1,
    int M, int K, int N, int relu) {
  __shared__ __align__(64) _Float16 lsA[128][32];
  __shared__ __align__(64) _Float16 lsB[64][32];

  const int tid    = threadIdx.x;
  const int lane   = tid & 31;
  const int wave   = tid >> 5;
  const int r16    = lane & 15;
  const int half   = lane >> 4;
  const int mBlock = blockIdx.x * 128;
  const int nBlock = blockIdx.y * 64;
  const int m0     = wave * 16;

  v8f acc[4] = {};

  for (int p = 0; p < nPairs; ++p) {
    const float*    A  = args.A[p];
    const _Float16* Bt = args.Bt[p];
    for (int k0 = 0; k0 < K; k0 += 32) {
      __syncthreads();
      {
        // A tile: 128 rows x 32 cols f32 -> f16 LDS; thread t: row=t>>1, 16 floats
        int row = tid >> 1;
        int col = (tid & 1) * 16;
        int grow = mBlock + row;
        if (grow < M) {
          const float* gp = A + (size_t)grow * K + k0 + col;
          if (k0 + 32 < K) __builtin_prefetch((const void*)(gp + 32), 0, 3);
          const float4* src4 = (const float4*)gp;
#pragma unroll
          for (int v = 0; v < 4; ++v) {
            float4 t = src4[v];
            lsA[row][col + 4 * v + 0] = (_Float16)t.x;
            lsA[row][col + 4 * v + 1] = (_Float16)t.y;
            lsA[row][col + 4 * v + 2] = (_Float16)t.z;
            lsA[row][col + 4 * v + 3] = (_Float16)t.w;
          }
        } else {
#pragma unroll
          for (int v = 0; v < 16; ++v) lsA[row][col + v] = (_Float16)0.0f;
        }
      }
      {
        // B tile: 64(N) x 32(K) f16 straight copy; thread t: row=t>>2, 8 halves
        int row = tid >> 2;
        int col = (tid & 3) * 8;
        *(uint4*)&lsB[row][col] =
            *(const uint4*)(Bt + (size_t)(nBlock + row) * K + k0 + col);
      }
      __syncthreads();

      v16h a = load_a_frag(&lsA[m0 + r16][0], half * 16);
#pragma unroll
      for (int j = 0; j < 4; ++j) {
        // B-fragment: lane half h holds 16 consecutive K (16h..16h+15) for its N column
        v16h b = *(const v16h*)&lsB[j * 16 + r16][half * 16];
        acc[j] = __builtin_amdgcn_wmma_f32_16x16x32_f16(
            false, a, false, b, (short)0, acc[j], false, false);
      }
    }
  }

#pragma unroll
  for (int j = 0; j < 4; ++j) {
    int col = nBlock + j * 16 + r16;
#pragma unroll
    for (int r = 0; r < 8; ++r) {
      int row = mBlock + m0 + r + half * 8;  // C layout: VGPR r -> M=r (lanes 0-15), M=r+8 (16-31)
      if (row < M) {
        float v = acc[j][r];
        if (bias0) v += bias0[col];
        if (bias1) v += bias1[col];
        if (relu)  v = fmaxf(v, 0.0f);
        C[(size_t)row * N + col] = v;
      }
    }
  }
}

// -------------------- host orchestration --------------------

static inline int cdiv_i(long long a, long long b) { return (int)((a + b - 1) / b); }

extern "C" void kernel_launch(void* const* d_in, const int* in_sizes, int n_in,
                              void* d_out, int out_size, void* d_ws, size_t ws_size,
                              hipStream_t stream) {
  (void)in_sizes; (void)n_in; (void)out_size; (void)ws_size;

  const float* x_paper  = (const float*)d_in[0];
  const float* x_author = (const float*)d_in[1];
  const float* l0c_Wl = (const float*)d_in[2];
  const float* l0c_b  = (const float*)d_in[3];
  const float* l0c_Wr = (const float*)d_in[4];
  const float* l0w_Wl = (const float*)d_in[5];
  const float* l0w_b  = (const float*)d_in[6];
  const float* l0w_Wr = (const float*)d_in[7];
  const float* l0r_Wl = (const float*)d_in[8];
  const float* l0r_b  = (const float*)d_in[9];
  const float* l0r_Wr = (const float*)d_in[10];
  const float* l1c_Wl = (const float*)d_in[11];
  const float* l1c_b  = (const float*)d_in[12];
  const float* l1c_Wr = (const float*)d_in[13];
  const float* l1w_Wl = (const float*)d_in[14];
  const float* l1w_b  = (const float*)d_in[15];
  const float* l1w_Wr = (const float*)d_in[16];
  // d_in[17..19]: l1_rev_* -- dead in the reference graph
  const float* lin_W  = (const float*)d_in[20];
  const float* lin_b  = (const float*)d_in[21];
  const int* ec1 = (const int*)d_in[22];
  const int* ec2 = (const int*)d_in[23];
  const int* ew1 = (const int*)d_in[24];
  const int* ew2 = (const int*)d_in[25];
  const int* er1 = (const int*)d_in[26];
  const int* er2 = (const int*)d_in[27];

  // ---- workspace layout (f32 element offsets) ----
  float* ws = (float*)d_ws;
  size_t o = 0;
  float* sum0_c = ws + o; o += (size_t)NP_H01C * F_IN;   // dead after phase C -> reused as l1 out
  float* sum0_w = ws + o; o += (size_t)NP_H01C * F_IN;
  float* sum0_r = ws + o; o += (size_t)NA_H01C * F_IN;
  float* cnt0_c = ws + o; o += NP_H01C;
  float* cnt0_w = ws + o; o += NP_H01C;
  float* cnt0_r = ws + o; o += NA_H01C;
  float* xp     = ws + o; o += (size_t)NP_H01C * HID;
  float* xa     = ws + o; o += (size_t)NA_H01C * HID;
  float* sum1_c = ws + o; o += (size_t)NP_H01C * HID;
  float* sum1_w = ws + o; o += (size_t)NP_H01C * HID;
  float* cnt1_c = ws + o; o += NP_H01C;
  float* cnt1_w = ws + o; o += NP_H01C;
  float* h1buf  = sum0_c;  // NP_H01C*256 floats, aliases sum0_c+sum0_w (same size, dead by then)

  // f16 transposed weights after the f32 region
  _Float16* h = (_Float16*)(ws + o);
  _Float16* wt_l0c_Wl = h; h += (size_t)F_IN * HID;
  _Float16* wt_l0c_Wr = h; h += (size_t)F_IN * HID;
  _Float16* wt_l0w_Wl = h; h += (size_t)F_IN * HID;
  _Float16* wt_l0w_Wr = h; h += (size_t)F_IN * HID;
  _Float16* wt_l0r_Wl = h; h += (size_t)F_IN * HID;
  _Float16* wt_l0r_Wr = h; h += (size_t)F_IN * HID;
  _Float16* wt_l1c_Wl = h; h += (size_t)HID * HID;
  _Float16* wt_l1c_Wr = h; h += (size_t)HID * HID;
  _Float16* wt_l1w_Wl = h; h += (size_t)HID * HID;
  _Float16* wt_l1w_Wr = h; h += (size_t)HID * HID;
  _Float16* wt_lin    = h; h += (size_t)HID * OUT_F;

  const dim3 blk(256);

  // ---- weight conversion ----
  {
    struct { const float* W; _Float16* Wt; int K, N; } cv[] = {
      {l0c_Wl, wt_l0c_Wl, F_IN, HID}, {l0c_Wr, wt_l0c_Wr, F_IN, HID},
      {l0w_Wl, wt_l0w_Wl, F_IN, HID}, {l0w_Wr, wt_l0w_Wr, F_IN, HID},
      {l0r_Wl, wt_l0r_Wl, F_IN, HID}, {l0r_Wr, wt_l0r_Wr, F_IN, HID},
      {l1c_Wl, wt_l1c_Wl, HID, HID},  {l1c_Wr, wt_l1c_Wr, HID, HID},
      {l1w_Wl, wt_l1w_Wl, HID, HID},  {l1w_Wr, wt_l1w_Wr, HID, HID},
      {lin_W,  wt_lin,    HID, OUT_F},
    };
    for (auto& c : cv)
      convert_weight_kernel<<<cdiv_i((long long)c.K * c.N, 256), blk, 0, stream>>>(
          c.W, c.Wt, c.K, c.N);
  }

  // ---- phase A: layer-0 aggregation (sums + counts) ----
  long long lenA = (long long)NP_H01C * F_IN * 2 + (long long)NA_H01C * F_IN +
                   2LL * NP_H01C + NA_H01C;  // sum0_c..cnt0_r contiguous
  fill0_kernel<<<2048, blk, 0, stream>>>(sum0_c, lenA);

  const int sBlocks = cdiv_i((long long)NEDGE * 32, 256);
  // cites: paper -> paper
  scatter_sum_kernel<<<sBlocks, blk, 0, stream>>>(x_paper, ec1, ec1 + NEDGE, NEDGE, F_IN, NP_H01C, sum0_c, cnt0_c);
  scatter_sum_kernel<<<sBlocks, blk, 0, stream>>>(x_paper, ec2, ec2 + NEDGE, NEDGE, F_IN, NP_H01C, sum0_c, cnt0_c);
  // writes: author -> paper
  scatter_sum_kernel<<<sBlocks, blk, 0, stream>>>(x_author, ew1, ew1 + NEDGE, NEDGE, F_IN, NP_H01C, sum0_w, cnt0_w);
  scatter_sum_kernel<<<sBlocks, blk, 0, stream>>>(x_author, ew2, ew2 + NEDGE, NEDGE, F_IN, NP_H01C, sum0_w, cnt0_w);
  // rev: paper -> author
  scatter_sum_kernel<<<sBlocks, blk, 0, stream>>>(x_paper, er1, er1 + NEDGE, NEDGE, F_IN, NA_H01C, sum0_r, cnt0_r);
  scatter_sum_kernel<<<sBlocks, blk, 0, stream>>>(x_paper, er2, er2 + NEDGE, NEDGE, F_IN, NA_H01C, sum0_r, cnt0_r);

  // ---- phase B: means ----
  mean_div_kernel<<<cdiv_i((long long)NP_H01C * F_IN, 256), blk, 0, stream>>>(sum0_c, cnt0_c, NP_H01C, F_IN);
  mean_div_kernel<<<cdiv_i((long long)NP_H01C * F_IN, 256), blk, 0, stream>>>(sum0_w, cnt0_w, NP_H01C, F_IN);
  mean_div_kernel<<<cdiv_i((long long)NA_H01C * F_IN, 256), blk, 0, stream>>>(sum0_r, cnt0_r, NA_H01C, F_IN);

  // ---- fused GEMM launcher: C = relu?(sum_p A_p@W_p + b0 + b1), single C write ----
  auto gemm = [&](const GemmArgs& ga, int nPairs, float* C, const float* b0,
                  const float* b1, int M, int K, int N, int relu) {
    dim3 g((unsigned)cdiv_i(M, 128), (unsigned)(N / 64));
    gemm_wmma_fused_kernel<<<g, blk, 0, stream>>>(ga, nPairs, C, b0, b1, M, K, N, relu);
  };

  // ---- phase C: layer-0, one kernel per node type ----
  {
    GemmArgs ga = {{sum0_c, x_paper, sum0_w, x_paper},
                   {wt_l0c_Wl, wt_l0c_Wr, wt_l0w_Wl, wt_l0w_Wr}};
    gemm(ga, 4, xp, l0c_b, l0w_b, NP_H01C, F_IN, HID, 1);
  }
  {
    GemmArgs ga = {{sum0_r, x_author, nullptr, nullptr},
                   {wt_l0r_Wl, wt_l0r_Wr, nullptr, nullptr}};
    gemm(ga, 2, xa, l0r_b, nullptr, NA_H01C, F_IN, HID, 1);
  }

  // ---- phase D: layer-1 aggregation ----
  long long lenB = (long long)NP_H01C * HID * 2 + 2LL * NP_H01C;  // sum1_c..cnt1_w contiguous
  fill0_kernel<<<2048, blk, 0, stream>>>(sum1_c, lenB);
  scatter_sum_kernel<<<sBlocks, blk, 0, stream>>>(xp, ec1, ec1 + NEDGE, NEDGE, HID, NP_H01C, sum1_c, cnt1_c);
  scatter_sum_kernel<<<sBlocks, blk, 0, stream>>>(xa, ew1, ew1 + NEDGE, NEDGE, HID, NP_H01C, sum1_w, cnt1_w);
  mean_div_kernel<<<cdiv_i((long long)NP_H01C * HID, 256), blk, 0, stream>>>(sum1_c, cnt1_c, NP_H01C, HID);
  mean_div_kernel<<<cdiv_i((long long)NP_H01C * HID, 256), blk, 0, stream>>>(sum1_w, cnt1_w, NP_H01C, HID);

  // ---- phase E: layer-1, single fused kernel into h1buf (aliases dead sum0_c/w) ----
  {
    GemmArgs ga = {{sum1_c, xp, sum1_w, xp},
                   {wt_l1c_Wl, wt_l1c_Wr, wt_l1w_Wl, wt_l1w_Wr}};
    gemm(ga, 4, h1buf, l1c_b, l1w_b, NP_H01C, HID, HID, 1);
  }

  // ---- phase F: final projection -> d_out (NP_H01 x 64, f32) ----
  {
    GemmArgs ga = {{h1buf, nullptr, nullptr, nullptr},
                   {wt_lin, nullptr, nullptr, nullptr}};
    gemm(ga, 1, (float*)d_out, lin_b, nullptr, NP_H01C, HID, OUT_F, 0);
  }
}
